// SAKE_modular_50818053046786
// MI455X (gfx1250) — matrix operand order repaired and checked by reading, not compiled, not run on hardware
//
#include <hip/hip_runtime.h>
#include <hip/hip_bf16.h>
#include <math.h>

// ---------------------------------------------------------------------------
// SAKE GNN on MI455X (gfx1250): B=512 graphs, M=32 nodes, H=128, L=2 layers.
// All dense GEMMs run on v_wmma_f32_16x16x32_f16 (f16 in, f32 accumulate).
// One workgroup (256 threads = 8 wave32) per graph; everything LDS-resident.
//   - A operand rows padded to 136 halves (272B) so the 16 row-lanes land on
//     disjoint 4-bank groups (no 16-way conflict at stride 256B).
//   - Weights pre-converted ONCE per launch to f16 in the swizzled LDS layout
//     [k/16][col][k%16] (prep kernel -> workspace, L2-resident). Per-block
//     weight staging is then a pure b128 global->LDS copy, with
//     global_prefetch of the next stage's slab overlapping the current GEMM.
// ---------------------------------------------------------------------------

typedef _Float16 half_t;
typedef half_t v16h __attribute__((ext_vector_type(16)));
typedef half_t v8h  __attribute__((ext_vector_type(8)));
typedef float  v8f  __attribute__((ext_vector_type(8)));

#define Bg 512
#define Mn 32
#define Hd 128
#define LDA 136        // padded leading dim (halves) for A operand tiles in LDS
#define WSLAB 16384    // halves per weight slab (128x128)

__device__ __forceinline__ float silu_f(float v) {
    return v / (1.0f + __expf(-v));
}

// swizzled index for f16 weight: tiles [k/16][col][k%16]
__device__ __forceinline__ int wsw_kb_col_kr(int kb, int col, int kr) {
    return (kb << 11) + (col << 4) + kr;
}

__device__ __forceinline__ v16h cat8(v8h lo, v8h hi) {
    return __builtin_shufflevector(lo, hi, 0, 1, 2, 3, 4, 5, 6, 7,
                                   8, 9, 10, 11, 12, 13, 14, 15);
}

// Wave-level 16x16 output tile GEMM: C += A[row_base:+16, 0:128] * W[0:128, col_base:+16]
// A: row-major [rows][LDA] f16 in LDS.  W: swizzled f16 in LDS (128x128 K x N).
// Fragment k order: contiguous k = 16*(lane/16) + s in both A and B (wave32).
__device__ __forceinline__ v8f wmma_tile(const half_t* A, const half_t* W,
                                         int row_base, int col_base, int lane, v8f acc) {
    // A 16x32 f16: lanes 0-15 row M, K 0..15; lanes 16-31 K 16..31
    const int row  = row_base + (lane & 15);
    const int aoff = (lane < 16) ? 0 : 16;
    // B 32x16 f16: lane holds column N=lane%16; lanes 0-15 K=0..15, lanes 16-31 K=16..31
    const int col  = col_base + (lane & 15);
    const int koff = (lane < 16) ? 0 : 16;
#pragma unroll
    for (int kk = 0; kk < Hd; kk += 32) {
        const half_t* p = A + row * LDA + kk + aoff;                    // 16B aligned
        v16h a = cat8(*(const v8h*)p, *(const v8h*)(p + 8));
        const half_t* q = W + (((kk + koff) >> 4) << 11) + (col << 4);  // 32B aligned
        v16h b = cat8(*(const v8h*)q, *(const v8h*)(q + 8));
        acc = __builtin_amdgcn_wmma_f32_16x16x32_f16(false, a, false, b, (short)0, acc,
                                                     false, false);
    }
    return acc;
}

// copy one pre-swizzled weight slab (32KB) into LDS; prefetch the next slab.
__device__ __forceinline__ void stage_weight(half_t* s_W, const half_t* slab,
                                             const half_t* next_slab, int tid) {
#pragma unroll
    for (int c = tid; c < 2048; c += 256)
        ((v8h*)s_W)[c] = ((const v8h*)slab)[c];        // global_load_b128 -> ds_store_b128
    if (next_slab)
        __builtin_prefetch(next_slab + tid * 64, 0, 1); // 256 lanes x 128B = 32KB
}

// ---------------------------------------------------------------------------
// Kernel 0: convert/swizzle all weight matrices to f16 once (14 slabs).
// slab: layer*6 + {0:Wa,1:Wb,2:We2,3:Wh1a,4:Wh1b,5:Wh2}; 12:W_out; 13:Wn1
// ---------------------------------------------------------------------------
__global__ __launch_bounds__(256) void sake_wprep_kernel(
    const float* __restrict__ We1, const float* __restrict__ We2,
    const float* __restrict__ Wh1, const float* __restrict__ Wh2,
    const float* __restrict__ W_out, const float* __restrict__ Wn1,
    half_t* __restrict__ wf16) {
    const int slab = blockIdx.x;
    const float* src;
    if (slab < 12) {
        int layer = slab / 6, s = slab % 6;
        switch (s) {
            case 0: src = We1 + layer * (2 * Hd + 1) * Hd; break;
            case 1: src = We1 + layer * (2 * Hd + 1) * Hd + Hd * Hd; break;
            case 2: src = We2 + layer * Hd * Hd; break;
            case 3: src = Wh1 + layer * 2 * Hd * Hd; break;
            case 4: src = Wh1 + layer * 2 * Hd * Hd + Hd * Hd; break;
            default: src = Wh2 + layer * Hd * Hd; break;
        }
    } else {
        src = (slab == 12) ? W_out : Wn1;
    }
    half_t* dst = wf16 + slab * WSLAB;
    const int tid = threadIdx.x;
    // destination-contiguous: d = (kb*128 + col)*16 + kr  ;  source = (kb*16+kr)*128 + col
    for (int c = tid; c < 2048; c += 256) {
        int base = c * 8;
        int kb = base >> 11, col = (base >> 4) & 127, kr0 = base & 15;
        v8h pk;
#pragma unroll
        for (int e = 0; e < 8; ++e)
            pk[e] = (half_t)src[(((kb << 4) + kr0 + e) << 7) + col];
        *(v8h*)(dst + base) = pk;
    }
}

// ---------------------------------------------------------------------------
// Kernel 1: per-graph geometry (d2) + input embedding hg0 = h @ W_in + b_in
// ---------------------------------------------------------------------------
__global__ __launch_bounds__(256) void sake_embed_kernel(
    const float* __restrict__ h, const float* __restrict__ x,
    const float* __restrict__ W_in, const float* __restrict__ b_in,
    float* __restrict__ hg, float* __restrict__ d2g) {
    const int g = blockIdx.x;
    const int tid = threadIdx.x;

    __shared__ float sx[Mn][3];
    for (int idx = tid; idx < Mn * 3; idx += 256) sx[idx / 3][idx % 3] = x[g * Mn * 3 + idx];
    __syncthreads();

    // pairwise squared distances (32x32)
    for (int idx = tid; idx < Mn * Mn; idx += 256) {
        int i = idx >> 5, j = idx & 31;
        float dx = sx[i][0] - sx[j][0];
        float dy = sx[i][1] - sx[j][1];
        float dz = sx[i][2] - sx[j][2];
        d2g[g * Mn * Mn + idx] = dx * dx + dy * dy + dz * dz;
    }

    // hg0: node m = tid>>3 computes 16 outputs starting at (tid&7)*16; F_IN=16
    int m = tid >> 3;
    int kb = (tid & 7) * 16;
    const float* hrow = h + (g * Mn + m) * 16;
    float hl[16];
#pragma unroll
    for (int f = 0; f < 16; ++f) hl[f] = hrow[f];
#pragma unroll
    for (int k = 0; k < 16; ++k) {
        int kk = kb + k;
        float acc = b_in[kk];
#pragma unroll
        for (int f = 0; f < 16; ++f) acc += hl[f] * W_in[f * Hd + kk];
        hg[(g * Mn + m) * Hd + kk] = acc;
    }
}

// ---------------------------------------------------------------------------
// Kernel 2: one SAKE interaction layer for one graph per block.
// ---------------------------------------------------------------------------
__global__ __launch_bounds__(256) void sake_layer_kernel(
    const float* __restrict__ We1, const float* __restrict__ be1,
    const float* __restrict__ be2, const float* __restrict__ bh1,
    const float* __restrict__ bh2, const half_t* __restrict__ wf16,
    float* __restrict__ hg, const float* __restrict__ d2g, int layer) {
    const int g = blockIdx.x;
    const int tid = threadIdx.x;
    const int wave = tid >> 5;
    const int lane = tid & 31;

    __shared__ float s_hg[Mn][Hd];      // 16 KB
    __shared__ float s_mi[Mn][Hd];      // 16 KB
    __shared__ float s_mj[Mn][Hd];      // 16 KB
    __shared__ float s_agg[Mn][Hd];     // 16 KB
    __shared__ float s_d2[Mn][Mn];      // 4 KB
    __shared__ float s_wd[Hd];
    __shared__ float s_b1[Hd];
    __shared__ float s_b2[Hd];
    __shared__ half_t s_A[128 * LDA];   // 34 KB: GEMM A operand (edge tiles / hg / agg / u)
    __shared__ half_t s_W[8 * 2048];    // 32 KB: current f16 weight, swizzled

    const half_t* slab = wf16 + layer * 6 * WSLAB;          // Wa Wb We2 Wh1a Wh1b Wh2
    const half_t* slab_after = wf16 + ((layer == 0) ? 6 : 12) * WSLAB;
    const float* We1l = We1 + layer * (2 * Hd + 1) * Hd;    // for wd row

    // ---- stage graph state ----
    for (int idx = tid; idx < Mn * Hd; idx += 256) s_hg[idx >> 7][idx & 127] = hg[g * Mn * Hd + idx];
    for (int idx = tid; idx < Mn * Mn; idx += 256) s_d2[idx >> 5][idx & 31] = d2g[g * Mn * Mn + idx];
    if (tid < Hd) {
        s_wd[tid] = We1l[2 * Hd * Hd + tid];   // We1[l, 2H, :]
        s_b1[tid] = be1[layer * Hd + tid];
        s_b2[tid] = be2[layer * Hd + tid];
    }

    // ---- mi = hg @ Wa ----
    for (int idx = tid; idx < Mn * 16; idx += 256) {      // vectorized f16 convert of hg
        int r = idx >> 4, k0 = (idx & 15) * 8;
        v8h pk;
#pragma unroll
        for (int k = 0; k < 8; ++k) pk[k] = (half_t)hg[g * Mn * Hd + r * Hd + k0 + k];
        *(v8h*)(s_A + r * LDA + k0) = pk;
    }
    stage_weight(s_W, slab + 0 * WSLAB, slab + 1 * WSLAB, tid);
    __syncthreads();
    for (int tt = 0; tt < 2; ++tt) {
        int t = wave * 2 + tt;
        int r = t >> 3, c = t & 7;
        v8f acc = {0.f, 0.f, 0.f, 0.f, 0.f, 0.f, 0.f, 0.f};
        acc = wmma_tile(s_A, s_W, r * 16, c * 16, lane, acc);
        int col = c * 16 + (lane & 15);
        int rb = r * 16 + ((lane < 16) ? 0 : 8);
#pragma unroll
        for (int v = 0; v < 8; ++v) s_mi[rb + v][col] = acc[v];
    }
    __syncthreads();

    // ---- mj = hg @ Wb ----
    stage_weight(s_W, slab + 1 * WSLAB, slab + 2 * WSLAB, tid);
    __syncthreads();
    for (int tt = 0; tt < 2; ++tt) {
        int t = wave * 2 + tt;
        int r = t >> 3, c = t & 7;
        v8f acc = {0.f, 0.f, 0.f, 0.f, 0.f, 0.f, 0.f, 0.f};
        acc = wmma_tile(s_A, s_W, r * 16, c * 16, lane, acc);
        int col = c * 16 + (lane & 15);
        int rb = r * 16 + ((lane < 16) ? 0 : 8);
#pragma unroll
        for (int v = 0; v < 8; ++v) s_mj[rb + v][col] = acc[v];
    }
    __syncthreads();

    // ---- edge MLP, tiled over (it: 2 x 16 i-rows) x (jt: 4 x 8 j-cols) ----
    for (int idx = tid; idx < Mn * Hd; idx += 256) s_agg[idx >> 7][idx & 127] = 0.0f;
    stage_weight(s_W, slab + 2 * WSLAB, slab + 3 * WSLAB, tid);
    __syncthreads();

    for (int it = 0; it < 2; ++it)
        for (int jt = 0; jt < 4; ++jt) {
            // build 128-edge tile: edge e = il*8+jl, il in [0,16), jl in [0,8)
            {
                int e = tid >> 1;
                int kb = (tid & 1) * 64;
                int il = e >> 3, jl = e & 7;
                int i = it * 16 + il, j = jt * 8 + jl;
                float dij = s_d2[i][j];
                for (int k0 = kb; k0 < kb + 64; k0 += 8) {
                    v8h pk;
#pragma unroll
                    for (int k = 0; k < 8; ++k) {
                        float pre = s_mi[i][k0 + k] + s_mj[j][k0 + k] +
                                    dij * s_wd[k0 + k] + s_b1[k0 + k];
                        pk[k] = (half_t)silu_f(pre);
                    }
                    *(v8h*)(s_A + e * LDA + k0) = pk;
                }
            }
            __syncthreads();
            // m2 = silu(m @ We2 + be2); masked aggregation over j into s_agg
            for (int tt = 0; tt < 8; ++tt) {
                int t = wave * 8 + tt;
                int r = t >> 3, c = t & 7;
                v8f acc = {0.f, 0.f, 0.f, 0.f, 0.f, 0.f, 0.f, 0.f};
                acc = wmma_tile(s_A, s_W, r * 16, c * 16, lane, acc);
                int col = c * 16 + (lane & 15);
                float bias = s_b2[col];
                int rb = (lane < 16) ? 0 : 8;
#pragma unroll
                for (int v = 0; v < 8; ++v) {
                    int row = r * 16 + rb + v;          // edge-row in [0,128)
                    int il = row >> 3, jl = row & 7;
                    int i = it * 16 + il, j = jt * 8 + jl;
                    float val = silu_f(acc[v] + bias);
                    if ((s_d2[i][j] < 1.0f) && (i != j))
                        atomicAdd(&s_agg[i][col], val);  // ds_add_f32
                }
            }
            __syncthreads();
        }

    // ---- u = silu(hg @ Wh1a + agg @ Wh1b + bh1) ----
    for (int idx = tid; idx < Mn * 16; idx += 256) {
        int r = idx >> 4, k0 = (idx & 15) * 8;
        v8h pk;
#pragma unroll
        for (int k = 0; k < 8; ++k) pk[k] = (half_t)s_hg[r][k0 + k];
        *(v8h*)(s_A + r * LDA + k0) = pk;
    }
    stage_weight(s_W, slab + 3 * WSLAB, slab + 4 * WSLAB, tid);
    __syncthreads();
    v8f uacc[2];
    for (int tt = 0; tt < 2; ++tt) {
        int t = wave * 2 + tt;
        int r = t >> 3, c = t & 7;
        v8f z = {0.f, 0.f, 0.f, 0.f, 0.f, 0.f, 0.f, 0.f};
        uacc[tt] = wmma_tile(s_A, s_W, r * 16, c * 16, lane, z);
    }
    __syncthreads();
    for (int idx = tid; idx < Mn * 16; idx += 256) {
        int r = idx >> 4, k0 = (idx & 15) * 8;
        v8h pk;
#pragma unroll
        for (int k = 0; k < 8; ++k) pk[k] = (half_t)s_agg[r][k0 + k];
        *(v8h*)(s_A + r * LDA + k0) = pk;
    }
    stage_weight(s_W, slab + 4 * WSLAB, slab + 5 * WSLAB, tid);
    __syncthreads();
    for (int tt = 0; tt < 2; ++tt) {
        int t = wave * 2 + tt;
        int r = t >> 3, c = t & 7;
        uacc[tt] = wmma_tile(s_A, s_W, r * 16, c * 16, lane, uacc[tt]);
        int col = c * 16 + (lane & 15);
        float b = bh1[layer * Hd + col];
        int rb = r * 16 + ((lane < 16) ? 0 : 8);
#pragma unroll
        for (int v = 0; v < 8; ++v) {
            float uv = silu_f(uacc[tt][v] + b);
            s_A[(Mn + rb + v) * LDA + col] = (half_t)uv;  // u in rows 32..63 (no overlap w/ A reads)
        }
    }
    __syncthreads();

    // ---- hg += u @ Wh2 + bh2 ----
    stage_weight(s_W, slab + 5 * WSLAB, slab_after, tid);
    __syncthreads();
    for (int tt = 0; tt < 2; ++tt) {
        int t = wave * 2 + tt;
        int r = t >> 3, c = t & 7;
        v8f z = {0.f, 0.f, 0.f, 0.f, 0.f, 0.f, 0.f, 0.f};
        v8f dacc = wmma_tile(s_A + Mn * LDA, s_W, r * 16, c * 16, lane, z);
        int col = c * 16 + (lane & 15);
        float b2 = bh2[layer * Hd + col];
        int rb = r * 16 + ((lane < 16) ? 0 : 8);
#pragma unroll
        for (int v = 0; v < 8; ++v) {
            int row = rb + v;
            hg[g * Mn * Hd + row * Hd + col] = s_hg[row][col] + dacc[v] + b2;
        }
    }
}

// ---------------------------------------------------------------------------
// Kernel 3: output head + per-graph reduction.
//   ho = hg@W_out + b_out; t = silu(ho@Wn1 + bn1); out[g] = sum_m (t.Wn2) + M*bn2
// ---------------------------------------------------------------------------
__global__ __launch_bounds__(256) void sake_head_kernel(
    const float* __restrict__ b_out, const float* __restrict__ bn1,
    const float* __restrict__ Wn2, const float* __restrict__ bn2,
    const half_t* __restrict__ wf16, const float* __restrict__ hg,
    float* __restrict__ out) {
    const int g = blockIdx.x;
    const int tid = threadIdx.x;
    const int wave = tid >> 5;
    const int lane = tid & 31;

    __shared__ half_t s_A[64 * LDA];    // rows 0..31: f16(hg); rows 32..63: f16(ho)
    __shared__ half_t s_W[8 * 2048];    // swizzled f16 weight
    __shared__ float s_t[Mn][Hd];
    __shared__ float s_sum;

    for (int idx = tid; idx < Mn * 16; idx += 256) {
        int r = idx >> 4, k0 = (idx & 15) * 8;
        v8h pk;
#pragma unroll
        for (int k = 0; k < 8; ++k) pk[k] = (half_t)hg[(g * Mn + r) * Hd + k0 + k];
        *(v8h*)(s_A + r * LDA + k0) = pk;
    }
    stage_weight(s_W, wf16 + 12 * WSLAB, wf16 + 13 * WSLAB, tid);
    if (tid == 0) s_sum = 0.0f;
    __syncthreads();

    // ho
    for (int tt = 0; tt < 2; ++tt) {
        int t = wave * 2 + tt;
        int r = t >> 3, c = t & 7;
        v8f acc = {0.f, 0.f, 0.f, 0.f, 0.f, 0.f, 0.f, 0.f};
        acc = wmma_tile(s_A, s_W, r * 16, c * 16, lane, acc);
        int col = c * 16 + (lane & 15);
        float b = b_out[col];
        int rb = r * 16 + ((lane < 16) ? 0 : 8);
#pragma unroll
        for (int v = 0; v < 8; ++v) s_A[(Mn + rb + v) * LDA + col] = (half_t)(acc[v] + b);
    }
    __syncthreads();
    stage_weight(s_W, wf16 + 13 * WSLAB, nullptr, tid);
    __syncthreads();

    // t = silu(ho @ Wn1 + bn1)
    for (int tt = 0; tt < 2; ++tt) {
        int t = wave * 2 + tt;
        int r = t >> 3, c = t & 7;
        v8f acc = {0.f, 0.f, 0.f, 0.f, 0.f, 0.f, 0.f, 0.f};
        acc = wmma_tile(s_A + Mn * LDA, s_W, r * 16, c * 16, lane, acc);
        int col = c * 16 + (lane & 15);
        float b = bn1[col];
        int rb = r * 16 + ((lane < 16) ? 0 : 8);
#pragma unroll
        for (int v = 0; v < 8; ++v) s_t[rb + v][col] = silu_f(acc[v] + b);
    }
    __syncthreads();

    float partial = 0.0f;
    for (int idx = tid; idx < Mn * Hd; idx += 256)
        partial += s_t[idx >> 7][idx & 127] * Wn2[idx & 127];
    atomicAdd(&s_sum, partial);
    __syncthreads();
    if (tid == 0) out[g] = s_sum + (float)Mn * bn2[0];
}

// ---------------------------------------------------------------------------
extern "C" void kernel_launch(void* const* d_in, const int* in_sizes, int n_in,
                              void* d_out, int out_size, void* d_ws, size_t ws_size,
                              hipStream_t stream) {
    (void)in_sizes; (void)n_in; (void)out_size; (void)ws_size;
    const float* h     = (const float*)d_in[0];
    const float* x     = (const float*)d_in[1];
    // d_in[2] = batch (int32): unused — nodes are contiguous per graph.
    const float* W_in  = (const float*)d_in[3];
    const float* b_in  = (const float*)d_in[4];
    const float* We1   = (const float*)d_in[5];
    const float* be1   = (const float*)d_in[6];
    const float* We2   = (const float*)d_in[7];
    const float* be2   = (const float*)d_in[8];
    const float* Wh1   = (const float*)d_in[9];
    const float* bh1   = (const float*)d_in[10];
    const float* Wh2   = (const float*)d_in[11];
    const float* bh2   = (const float*)d_in[12];
    const float* W_out = (const float*)d_in[13];
    const float* b_out = (const float*)d_in[14];
    const float* Wn1   = (const float*)d_in[15];
    const float* bn1   = (const float*)d_in[16];
    const float* Wn2   = (const float*)d_in[17];
    const float* bn2   = (const float*)d_in[18];
    float* out = (float*)d_out;

    // workspace: hg (8MB) | d2 (2MB) | wf16 (14 slabs * 32KB = 448KB)
    float* hg   = (float*)d_ws;
    float* d2g  = hg + (size_t)Bg * Mn * Hd;
    half_t* wf16 = (half_t*)(d2g + (size_t)Bg * Mn * Mn);

    sake_wprep_kernel<<<14, 256, 0, stream>>>(We1, We2, Wh1, Wh2, W_out, Wn1, wf16);
    sake_embed_kernel<<<Bg, 256, 0, stream>>>(h, x, W_in, b_in, hg, d2g);
    sake_layer_kernel<<<Bg, 256, 0, stream>>>(We1, be1, be2, bh1, bh2, wf16, hg, d2g, 0);
    sake_layer_kernel<<<Bg, 256, 0, stream>>>(We1, be1, be2, bh1, bh2, wf16, hg, d2g, 1);
    sake_head_kernel<<<Bg, 256, 0, stream>>>(b_out, bn1, Wn2, bn2, wf16, hg, out);
}
